// CASchNetEncoder_44848048505536
// MI455X (gfx1250) — matrix operand order repaired and checked by reading, not compiled, not run on hardware
//
#include <hip/hip_runtime.h>

// ---------------------------------------------------------------------------
// CASchNet encoder for MI455X (gfx1250, wave32, WMMA f32<-f16).
//
// Roofline: edge MLP dominates compute (2.2e11 FLOPs); HBM floor ~150-200us
// at 23.3 TB/s with f16 edge features. The naive scatter costs 492M f32
// atomics -> likely the real bottleneck. So: counting-sort edges by dst once
// per call, process 64 sorted edges per block iteration, and do an LDS
// segmented column reduction before touching agg (~10x fewer atomics, all
// L2-resident). All GEMMs: V_WMMA_F32_16X16X32_F16, weights staged in LDS,
// B fragments reused across 4 M-tiles per K-step.
// ---------------------------------------------------------------------------

#define N_NODES   40000
#define E_EDGES   640000
#define H_DIM     128
#define FLT_DIM   128
#define EC_DIM    100
#define L_LAYERS  6
#define IN_DIM    5
#define ZROW      (IN_DIM + H_DIM)   // 133
#define CUTOFF_F  10.0f
#define LOG2_F    0.6931471805599453f

typedef __attribute__((ext_vector_type(16))) _Float16 v16h;
typedef __attribute__((ext_vector_type(8)))  float    v8f;

__device__ __forceinline__ float ssp(float x) {
  float ax = fabsf(x);
  return fmaxf(x, 0.0f) + log1pf(expf(-ax)) - LOG2_F;
}

__device__ __forceinline__ v8f wmma_f16(v16h a, v16h b, v8f c) {
  return __builtin_amdgcn_wmma_f32_16x16x32_f16(
      false, a, false, b, (short)0, c, false, false);
}

// A fragment: 16x32 f16 tile from row-major [16][ld] f16.
// lane l -> row m=l&15; lanes 0-15 hold K=0..15, lanes 16-31 K=16..31.
__device__ __forceinline__ v16h load_A16(const _Float16* base, int lane,
                                         int kbase, int ld) {
  const int m  = lane & 15;
  const int kh = (lane >> 4) & 1;
  const _Float16* row = base + m * ld + kbase + kh * 16;
  v16h a;
#pragma unroll
  for (int v = 0; v < 8; ++v) {
    int k0 = 2 * v;
    a[2 * v]     = row[k0];
    a[2 * v + 1] = row[k0 + 1];
  }
  return a;
}

// B fragment: B[k][n] = W[n][k], W row-major [n][ld] f16.
// lane l -> col n=(l&15)+nbase; lanes 0-15 K=0..15, lanes 16-31 K=16..31.
// 32 contiguous bytes per lane -> 2x b128 LDS loads.
__device__ __forceinline__ v16h load_Bw(const _Float16* w, int lane,
                                        int kbase, int ld, int nbase) {
  const int n  = nbase + (lane & 15);
  const int kb = (lane >> 4) & 1;
  const _Float16* row = w + n * ld + kbase + kb * 16;
  v16h b;
#pragma unroll
  for (int v = 0; v < 8; ++v) {
    b[2 * v]     = row[2 * v];
    b[2 * v + 1] = row[2 * v + 1];
  }
  return b;
}

// ---------------------------------------------------------------------------
// Counting sort of edges by dst
// ---------------------------------------------------------------------------

__global__ void hist_kernel(const int* __restrict__ dst, int* __restrict__ hist) {
  int e = blockIdx.x * blockDim.x + threadIdx.x;
  if (e < E_EDGES) atomicAdd(&hist[dst[e]], 1);
}

// single-block exclusive scan of hist (40000 bins) -> cursor
__global__ __launch_bounds__(1024)
void scan_kernel(const int* __restrict__ hist, int* __restrict__ cursor) {
  __shared__ int tmp[1024];
  __shared__ int carry;
  const int t = threadIdx.x;
  if (t == 0) carry = 0;
  __syncthreads();
  for (int base = 0; base < N_NODES; base += 1024) {
    int idx = base + t;
    int v = (idx < N_NODES) ? hist[idx] : 0;
    tmp[t] = v;
    __syncthreads();
    for (int off = 1; off < 1024; off <<= 1) {
      int x = (t >= off) ? tmp[t - off] : 0;
      __syncthreads();
      tmp[t] += x;
      __syncthreads();
    }
    int incl  = tmp[t];
    int total = tmp[1023];
    int cprev = carry;
    __syncthreads();
    if (t == 0) carry = cprev + total;
    if (idx < N_NODES) cursor[idx] = cprev + incl - v;
    __syncthreads();
  }
}

__global__ void scatter_kernel(const int* __restrict__ src,
                               const int* __restrict__ dst,
                               const float* __restrict__ elen,
                               int* __restrict__ cursor,
                               int* __restrict__ perm,
                               int* __restrict__ src_s,
                               int* __restrict__ dst_s,
                               float* __restrict__ cut_s) {
  int e = blockIdx.x * blockDim.x + threadIdx.x;
  if (e >= E_EDGES) return;
  int d = dst[e];
  int pos = atomicAdd(&cursor[d], 1);
  perm[pos]  = e;
  dst_s[pos] = d;
  src_s[pos] = src[e];
  cut_s[pos] = (elen[e] <= CUTOFF_F) ? 1.0f : 0.0f;
}

// edge_attr -> f16, padded 100->128, gathered into sorted order (one-time)
__global__ void convert_edge_attr_sorted_kernel(const float* __restrict__ ea,
                                                const int* __restrict__ perm,
                                                _Float16* __restrict__ out) {
  size_t i = (size_t)blockIdx.x * blockDim.x + threadIdx.x;  // over E*128
  if (i >= (size_t)E_EDGES * 128) return;
  int k = (int)(i & 127);
  size_t p = i >> 7;
  int e = perm[p];
  out[i] = (k < EC_DIM) ? (_Float16)ea[(size_t)e * EC_DIM + k] : (_Float16)0.0f;
}

// dest layout: [l][mat][n(out)][k(in)] f16; mat: 0=nW1(pad),1=nW2,2=lin1,3=lin2,4=lin
__global__ void convert_weights_kernel(const float* __restrict__ nW1,
                                       const float* __restrict__ nW2,
                                       const float* __restrict__ l1W,
                                       const float* __restrict__ l2W,
                                       const float* __restrict__ lW,
                                       _Float16* __restrict__ out) {
  int i = blockIdx.x * blockDim.x + threadIdx.x;  // over L*5*128*128
  if (i >= L_LAYERS * 5 * 128 * 128) return;
  int k   = i & 127;
  int n   = (i >> 7) & 127;
  int mat = (i >> 14) % 5;
  int l   = (i >> 14) / 5;
  float v;
  switch (mat) {
    case 0:  v = (k < EC_DIM) ? nW1[((size_t)l * 128 + n) * EC_DIM + k] : 0.0f; break;
    case 1:  v = nW2[((size_t)l * 128 + n) * 128 + k]; break;
    case 2:  v = l1W[((size_t)l * 128 + n) * 128 + k]; break;
    case 3:  v = l2W[((size_t)l * 128 + n) * 128 + k]; break;
    default: v = lW [((size_t)l * 128 + n) * 128 + k]; break;
  }
  out[i] = (_Float16)v;
}

// h = feats @ emblin_W.T + emblin_b + ptemb
__global__ void embed_kernel(const float* __restrict__ z,
                             const float* __restrict__ eW,
                             const float* __restrict__ eb,
                             float* __restrict__ h) {
  size_t i = (size_t)blockIdx.x * blockDim.x + threadIdx.x;
  if (i >= (size_t)N_NODES * H_DIM) return;
  int j = (int)(i & 127);
  size_t node = i >> 7;
  const float* zr = z + node * ZROW;
  float s = eb[j] + zr[IN_DIM + j];
#pragma unroll
  for (int k = 0; k < IN_DIM; ++k) s += zr[k] * eW[j * IN_DIM + k];
  h[i] = s;
}

// ---------------------------------------------------------------------------
// xsrc = h @ l1W.T ; 64 nodes per block iteration (4 M-tiles, B reuse)
// ---------------------------------------------------------------------------
__global__ __launch_bounds__(256)
void node_lin1_kernel(const float* __restrict__ h,
                      const _Float16* __restrict__ w,
                      float* __restrict__ xsrc) {
  __shared__ _Float16 sW[128 * 128];
  __shared__ _Float16 sH[64 * 128];
  const int tid = threadIdx.x, wave = tid >> 5, lane = tid & 31;
  for (int i = tid; i < 128 * 128 / 2; i += 256)
    ((unsigned int*)sW)[i] = ((const unsigned int*)w)[i];
  const int nbase = wave * 16;
  const int n = nbase + (lane & 15);
  const int ntiles = N_NODES / 64;
  for (int tile = blockIdx.x; tile < ntiles; tile += gridDim.x) {
    __syncthreads();
    const size_t i0 = (size_t)tile * 64;
    for (int i = tid; i < 64 * 128; i += 256)
      sH[i] = (_Float16)h[i0 * 128 + i];
    __syncthreads();
    v8f acc[4] = {};
#pragma unroll
    for (int kt = 0; kt < 4; ++kt) {
      v16h b = load_Bw(sW, lane, kt * 32, 128, nbase);
#pragma unroll
      for (int mt = 0; mt < 4; ++mt) {
        v16h a = load_A16(sH + mt * 16 * 128, lane, kt * 32, 128);
        acc[mt] = wmma_f16(a, b, acc[mt]);
      }
    }
#pragma unroll
    for (int mt = 0; mt < 4; ++mt)
#pragma unroll
      for (int r = 0; r < 8; ++r) {
        int m = mt * 16 + r + 8 * (lane >> 4);
        xsrc[(i0 + m) * 128 + n] = acc[mt][r];
      }
  }
}

// ---------------------------------------------------------------------------
// Fused edge kernel over dst-sorted edges, 64 edges per block iteration:
//   filt = (ssp(ea@W1.T + b1) @ W2.T + b2) * cut
//   msg  = filt * xsrc[src]   -> LDS
//   segmented column reduction over sorted dst -> few atomics into agg
// ---------------------------------------------------------------------------
__global__ __launch_bounds__(256)
void edge_filter_kernel(const _Float16* __restrict__ eaF16,  // (E,128) sorted
                        const float* __restrict__ xsrc,
                        const int* __restrict__ src_s,
                        const int* __restrict__ dst_s,
                        const float* __restrict__ cut_s,
                        const _Float16* __restrict__ w1,
                        const _Float16* __restrict__ w2,
                        const float* __restrict__ b1,
                        const float* __restrict__ b2,
                        float* __restrict__ agg) {
  __shared__ _Float16 sW1[128 * 128];
  __shared__ _Float16 sW2[128 * 128];
  // phase-aliased 32KB region:
  //   phase 1: sEA (64x128 f16, 16KB) + sInter (64x128 f16, 16KB)
  //   phase 2: sMsg (64x128 f32, 32KB)
  __shared__ __align__(16) char sRegion[64 * 128 * 4];
  __shared__ int   sSrc[64];
  __shared__ int   sDst[64];
  __shared__ float sCut[64];

  _Float16* sEA    = (_Float16*)sRegion;
  _Float16* sInter = (_Float16*)(sRegion + 64 * 128 * 2);
  float*    sMsg   = (float*)sRegion;

  const int tid = threadIdx.x, wave = tid >> 5, lane = tid & 31;

  for (int i = tid; i < 128 * 128 / 2; i += 256) {
    ((unsigned int*)sW1)[i] = ((const unsigned int*)w1)[i];
    ((unsigned int*)sW2)[i] = ((const unsigned int*)w2)[i];
  }

  const int nbase = wave * 16;
  const int n = nbase + (lane & 15);
  const float bias1 = b1[n];
  const float bias2 = b2[n];

  const int ntiles = E_EDGES / 64;
  for (int tile = blockIdx.x; tile < ntiles; tile += gridDim.x) {
    __syncthreads();  // previous iteration's reduce done; weights staged (1st)
    const int e0 = tile * 64;
    // 64 rows x 256B = 16KB: 4x b128 per thread, coalesced
#pragma unroll
    for (int q = 0; q < 4; ++q)
      ((uint4*)sEA)[tid + q * 256] =
          ((const uint4*)(eaF16 + (size_t)e0 * 128))[tid + q * 256];
    if (tid < 64) {
      sSrc[tid] = src_s[e0 + tid];
      sDst[tid] = dst_s[e0 + tid];
      sCut[tid] = cut_s[e0 + tid];
    }
    __syncthreads();

    // GEMM1: 64x128 @ W1.T, B fragment reused across 4 M-tiles
    v8f acc[4] = {};
#pragma unroll
    for (int kt = 0; kt < 4; ++kt) {
      v16h b = load_Bw(sW1, lane, kt * 32, 128, nbase);
#pragma unroll
      for (int mt = 0; mt < 4; ++mt) {
        v16h a = load_A16(sEA + mt * 16 * 128, lane, kt * 32, 128);
        acc[mt] = wmma_f16(a, b, acc[mt]);
      }
    }
#pragma unroll
    for (int mt = 0; mt < 4; ++mt)
#pragma unroll
      for (int r = 0; r < 8; ++r) {
        int m = mt * 16 + r + 8 * (lane >> 4);
        sInter[m * 128 + n] = (_Float16)ssp(acc[mt][r] + bias1);
      }
    __syncthreads();

    // GEMM2: inter @ W2.T
    v8f acc2[4] = {};
#pragma unroll
    for (int kt = 0; kt < 4; ++kt) {
      v16h b = load_Bw(sW2, lane, kt * 32, 128, nbase);
#pragma unroll
      for (int mt = 0; mt < 4; ++mt) {
        v16h a = load_A16(sInter + mt * 16 * 128, lane, kt * 32, 128);
        acc2[mt] = wmma_f16(a, b, acc2[mt]);
      }
    }
    __syncthreads();  // sEA/sInter dead -> region becomes sMsg

    // msg = filt * cut * gather(xsrc)
#pragma unroll
    for (int mt = 0; mt < 4; ++mt)
#pragma unroll
      for (int r = 0; r < 8; ++r) {
        int m = mt * 16 + r + 8 * (lane >> 4);
        float wv = acc2[mt][r] + bias2;
        float xv = xsrc[(size_t)sSrc[m] * H_DIM + n];
        sMsg[m * 128 + n] = wv * xv * sCut[m];
      }
    __syncthreads();

    // segmented column reduction: thread -> (col, 32-row half); dst-sorted so
    // runs are long (avg degree 16) -> ~1 atomic per (col, distinct dst)
    {
      const int col = tid & 127;
      const int r0  = (tid >> 7) * 32;
      float run = 0.0f;
      int curd = sDst[r0];
      for (int r = r0; r < r0 + 32; ++r) {
        int d = sDst[r];
        if (d != curd) {
          atomicAdd(&agg[(size_t)curd * H_DIM + col], run);
          run = 0.0f;
          curd = d;
        }
        run += sMsg[r * 128 + col];
      }
      atomicAdd(&agg[(size_t)curd * H_DIM + col], run);
    }
  }
}

// ---------------------------------------------------------------------------
// h += ( ssp(agg @ l2W.T + l2b) @ lW.T + lb ) ; 64 nodes per iteration
// ---------------------------------------------------------------------------
__global__ __launch_bounds__(256)
void node_update_kernel(const float* __restrict__ agg,
                        const _Float16* __restrict__ w2,
                        const float* __restrict__ b2,
                        const _Float16* __restrict__ wl,
                        const float* __restrict__ bl,
                        float* __restrict__ h) {
  __shared__ _Float16 sW2[128 * 128];
  __shared__ _Float16 sWl[128 * 128];
  __shared__ _Float16 sA[64 * 128];
  __shared__ _Float16 sInter[64 * 128];
  const int tid = threadIdx.x, wave = tid >> 5, lane = tid & 31;

  for (int i = tid; i < 128 * 128 / 2; i += 256) {
    ((unsigned int*)sW2)[i] = ((const unsigned int*)w2)[i];
    ((unsigned int*)sWl)[i] = ((const unsigned int*)wl)[i];
  }

  const int nbase = wave * 16;
  const int n = nbase + (lane & 15);
  const float bias2 = b2[n];
  const float biasl = bl[n];

  const int ntiles = N_NODES / 64;
  for (int tile = blockIdx.x; tile < ntiles; tile += gridDim.x) {
    __syncthreads();
    const size_t i0 = (size_t)tile * 64;
    for (int i = tid; i < 64 * 128; i += 256)
      sA[i] = (_Float16)agg[i0 * 128 + i];
    __syncthreads();

    v8f acc[4] = {};
#pragma unroll
    for (int kt = 0; kt < 4; ++kt) {
      v16h b = load_Bw(sW2, lane, kt * 32, 128, nbase);
#pragma unroll
      for (int mt = 0; mt < 4; ++mt) {
        v16h a = load_A16(sA + mt * 16 * 128, lane, kt * 32, 128);
        acc[mt] = wmma_f16(a, b, acc[mt]);
      }
    }
#pragma unroll
    for (int mt = 0; mt < 4; ++mt)
#pragma unroll
      for (int r = 0; r < 8; ++r) {
        int m = mt * 16 + r + 8 * (lane >> 4);
        sInter[m * 128 + n] = (_Float16)ssp(acc[mt][r] + bias2);
      }
    __syncthreads();

    v8f acc2[4] = {};
#pragma unroll
    for (int kt = 0; kt < 4; ++kt) {
      v16h b = load_Bw(sWl, lane, kt * 32, 128, nbase);
#pragma unroll
      for (int mt = 0; mt < 4; ++mt) {
        v16h a = load_A16(sInter + mt * 16 * 128, lane, kt * 32, 128);
        acc2[mt] = wmma_f16(a, b, acc2[mt]);
      }
    }
#pragma unroll
    for (int mt = 0; mt < 4; ++mt)
#pragma unroll
      for (int r = 0; r < 8; ++r) {
        int m = mt * 16 + r + 8 * (lane >> 4);
        size_t idx = (i0 + m) * 128 + n;
        h[idx] = h[idx] + (acc2[mt][r] + biasl);  // residual
      }
  }
}

// ---------------------------------------------------------------------------
extern "C" void kernel_launch(void* const* d_in, const int* in_sizes, int n_in,
                              void* d_out, int out_size, void* d_ws, size_t ws_size,
                              hipStream_t stream) {
  const float* z     = (const float*)d_in[0];
  // d_in[1] = p_ctx: unused by the reference
  const int*   eidx  = (const int*)d_in[2];
  const float* elen  = (const float*)d_in[3];
  const float* eattr = (const float*)d_in[4];
  const float* embW  = (const float*)d_in[5];
  const float* embB  = (const float*)d_in[6];
  const float* l1W   = (const float*)d_in[7];
  const float* nW1   = (const float*)d_in[8];
  const float* nb1   = (const float*)d_in[9];
  const float* nW2   = (const float*)d_in[10];
  const float* nb2   = (const float*)d_in[11];
  const float* l2W   = (const float*)d_in[12];
  const float* l2b   = (const float*)d_in[13];
  const float* lW    = (const float*)d_in[14];
  const float* lb    = (const float*)d_in[15];

  float* h = (float*)d_out;  // running h buffer (N,128) f32

  // workspace layout (all offsets 256B-aligned)
  char* ws = (char*)d_ws;
  size_t off = 0;
  _Float16* eaF16 = (_Float16*)(ws + off); off += (size_t)E_EDGES * 128 * 2;
  float* xsrc     = (float*)(ws + off);    off += (size_t)N_NODES * H_DIM * 4;
  float* agg      = (float*)(ws + off);    off += (size_t)N_NODES * H_DIM * 4;
  _Float16* wF16  = (_Float16*)(ws + off); off += (size_t)L_LAYERS * 5 * 16384 * 2;
  int* hist       = (int*)(ws + off);      off += (size_t)N_NODES * 4;
  int* cursor     = (int*)(ws + off);      off += (size_t)N_NODES * 4;
  int* perm       = (int*)(ws + off);      off += (size_t)E_EDGES * 4;
  int* src_s      = (int*)(ws + off);      off += (size_t)E_EDGES * 4;
  int* dst_s      = (int*)(ws + off);      off += (size_t)E_EDGES * 4;
  float* cut_s    = (float*)(ws + off);    off += (size_t)E_EDGES * 4;

  const int* src = eidx;
  const int* dstp = eidx + E_EDGES;

  // ---- counting sort of edges by dst ----
  hipMemsetAsync(hist, 0, (size_t)N_NODES * 4, stream);
  hist_kernel<<<(E_EDGES + 255) / 256, 256, 0, stream>>>(dstp, hist);
  scan_kernel<<<1, 1024, 0, stream>>>(hist, cursor);
  scatter_kernel<<<(E_EDGES + 255) / 256, 256, 0, stream>>>(
      src, dstp, elen, cursor, perm, src_s, dst_s, cut_s);
  convert_edge_attr_sorted_kernel<<<(unsigned)(((size_t)E_EDGES * 128 + 255) / 256),
                                    256, 0, stream>>>(eattr, perm, eaF16);

  // ---- one-time conversions + embedding ----
  convert_weights_kernel<<<(L_LAYERS * 5 * 16384 + 255) / 256, 256, 0, stream>>>(
      nW1, nW2, l1W, l2W, lW, wF16);
  embed_kernel<<<(N_NODES * H_DIM + 255) / 256, 256, 0, stream>>>(z, embW, embB, h);

  // ---- layers ----
  for (int l = 0; l < L_LAYERS; ++l) {
    const _Float16* wn1 = wF16 + ((size_t)l * 5 + 0) * 16384;
    const _Float16* wn2 = wF16 + ((size_t)l * 5 + 1) * 16384;
    const _Float16* wl1 = wF16 + ((size_t)l * 5 + 2) * 16384;
    const _Float16* wl2 = wF16 + ((size_t)l * 5 + 3) * 16384;
    const _Float16* wll = wF16 + ((size_t)l * 5 + 4) * 16384;

    node_lin1_kernel<<<625, 256, 0, stream>>>(h, wl1, xsrc);
    hipMemsetAsync(agg, 0, (size_t)N_NODES * H_DIM * 4, stream);
    edge_filter_kernel<<<1600, 256, 0, stream>>>(
        eaF16, xsrc, src_s, dst_s, cut_s, wn1, wn2,
        nb1 + l * FLT_DIM, nb2 + l * FLT_DIM, agg);
    node_update_kernel<<<625, 256, 0, stream>>>(
        agg, wl2, l2b + l * H_DIM, wll, lb + l * H_DIM, h);
  }
}
